// ResidualGNN_52097953300757
// MI455X (gfx1250) — matrix operand order repaired and checked by reading, not compiled, not run on hardware
//
#include <hip/hip_runtime.h>
#include <hip/hip_bf16.h>
#include <math.h>

// ---------------------------------------------------------------- constants
#define NG        512        // graphs
#define FN        256        // nodes per graph == node feature dim
#define NN        131072     // total nodes
#define HC        32         // hidden channels
#define NE        4194304    // edges
#define TRIU      32640      // FN*(FN-1)/2  (== 1020 chunks of 32: chunk-aligned!)
#define IN1       32736      // TRIU + 3*HC
#define HID       64
#define EPS       1e-5f
#define LDA       40         // LDS row stride in bf16 (80B = 20 banks, 16B-aligned)

typedef __bf16 bf16x16 __attribute__((ext_vector_type(16)));
typedef float  f32x8   __attribute__((ext_vector_type(8)));

union Frag {                 // one WMMA bf16 operand = two 16B LDS beats
  bf16x16 v;
  uint4   q[2];
};

__device__ static inline unsigned pack2(float a, float b) {
  union { __bf16 h[2]; unsigned u; } p;
  p.h[0] = (__bf16)a; p.h[1] = (__bf16)b;
  return p.u;
}

// ---- CDNA5 async global->LDS copy (ASYNCcnt-tracked, no VGPR data path)
__device__ static inline unsigned lds_addr(const void* p) {
  return (unsigned)(unsigned long long)p;   // low 32 bits of LDS aperture = offset
}
__device__ static inline void async_copy_b128(unsigned lds_off, const void* gptr) {
  asm volatile("global_load_async_to_lds_b128 %0, %1, off"
               :: "v"(lds_off), "v"((unsigned long long)gptr) : "memory");
}
__device__ static inline void wait_async0() {
  asm volatile("s_wait_asynccnt 0x0" ::: "memory");
}

// ---------------------------------------------------------------- utilities
__global__ void k_fill4(float4* __restrict__ p, float v, int n4) {
  int i = blockIdx.x * blockDim.x + threadIdx.x;
  if (i < n4) p[i] = make_float4(v, v, v, v);
}

__global__ void k_deg(const int* __restrict__ dst, float* __restrict__ deg) {
  int i = blockIdx.x * blockDim.x + threadIdx.x;
  if (i < NE) atomicAdd(&deg[dst[i]], 1.0f);
}

__global__ void k_rsqrt_ip(float* __restrict__ p, int n) {
  int i = blockIdx.x * blockDim.x + threadIdx.x;
  if (i < n) p[i] = rsqrtf(p[i]);
}

// -------- conv weight prep: Wt[col][k] = bf16(W[k][col]),  Wt: [32 x K]
__global__ void k_wconv(const float* __restrict__ W, __bf16* __restrict__ Wt, int K) {
  int i = blockIdx.x * blockDim.x + threadIdx.x;
  if (i >= 32 * K) return;
  int cc = i / K, k = i - cc * K;
  Wt[i] = (__bf16)W[(size_t)k * 32 + cc];
}

// ---------------------------------------------------- GCN GEMM: H = A @ W
// A: [NN,K] (f32 x for layer 0, bf16 cur for layers 1-2), Wt: [32][K] bf16.
// Double-buffered LDS; async DMA for bf16 operands overlaps WMMA.
template <bool ASYNC_A>
__global__ __launch_bounds__(256) void k_gcn_gemm(const float* __restrict__ A,
                                                  const __bf16* __restrict__ Abf,
                                                  const __bf16* __restrict__ Wt,
                                                  float* __restrict__ H, int K) {
  __shared__ __attribute__((aligned(16))) __bf16 As[2 * 128 * LDA];
  __shared__ __attribute__((aligned(16))) __bf16 Bst[2 * 32 * LDA];  // [col][k]
  const int tid  = threadIdx.x;
  const int wave = tid >> 5;           // gfx1250 wave32
  const int lane = tid & 31;
  const int half = lane >> 4;
  const int mrow = lane & 15;
  const int rowBase = blockIdx.x * 128;

  auto stage = [&](int kc, int buf) {
    __bf16* as = &As[buf * 128 * LDA];
    __bf16* bs = &Bst[buf * 32 * LDA];
    if (ASYNC_A) {
      for (int i = tid; i < 512; i += 256) {          // 128x32 bf16, 16B beats
        int r = i >> 2, kq = (i & 3) * 8;
        async_copy_b128(lds_addr(&as[r * LDA + kq]),
                        &Abf[(size_t)(rowBase + r) * K + kc + kq]);
      }
    } else {
      for (int i = tid; i < 128 * 32 / 4; i += 256) { // f32 -> cvt_pk_bf16
        int r = i >> 3, kq = (i & 7) * 4;
        const float4 f4 = *(const float4*)&A[(size_t)(rowBase + r) * K + kc + kq];
        uint2 u; u.x = pack2(f4.x, f4.y); u.y = pack2(f4.z, f4.w);
        *(uint2*)&as[r * LDA + kq] = u;
      }
    }
    if (tid < 128) {                                  // 32 cols x 32 k bf16
      int cc = tid >> 2, rq = (tid & 3) * 8;
      async_copy_b128(lds_addr(&bs[cc * LDA + rq]),
                      &Wt[(size_t)cc * K + kc + rq]);
    }
  };

  f32x8 c0 = {}, c1 = {};
  const int nc = K / 32;
  stage(0, 0);
  for (int c = 0; c < nc; ++c) {
    wait_async0();
    __syncthreads();                      // buf c&1 ready (async + ds stores)
    if (c + 1 < nc) stage((c + 1) * 32, (c + 1) & 1); // issue-only: overlaps WMMA
    const __bf16* as = &As[(c & 1) * 128 * LDA];
    const __bf16* bs = &Bst[(c & 1) * 32 * LDA];

    Frag af, b0f, b1f;
    const int rb = (wave * 16 + mrow) * LDA;
    const int kb = half * 8;
    af.q[0] = *(const uint4*)&as[rb + kb];
    af.q[1] = *(const uint4*)&as[rb + 16 + kb];
    b0f.q[0] = *(const uint4*)&bs[mrow * LDA + half * 8];
    b0f.q[1] = *(const uint4*)&bs[mrow * LDA + half * 8 + 16];
    b1f.q[0] = *(const uint4*)&bs[(16 + mrow) * LDA + half * 8];
    b1f.q[1] = *(const uint4*)&bs[(16 + mrow) * LDA + half * 8 + 16];

    c0 = __builtin_amdgcn_wmma_f32_16x16x32_bf16(false, af.v, false, b0f.v,
                                                 (short)0, c0, false, false);
    c1 = __builtin_amdgcn_wmma_f32_16x16x32_bf16(false, af.v, false, b1f.v,
                                                 (short)0, c1, false, false);
  }
  const int gr = rowBase + wave * 16 + half * 8;   // C: c[v] = (M=v+8*half, N=mrow)
#pragma unroll
  for (int v = 0; v < 8; ++v) {
    H[(size_t)(gr + v) * 32 + mrow]      = c0[v];
    H[(size_t)(gr + v) * 32 + 16 + mrow] = c1[v];
  }
}

// ------------------------------------------------- edge scatter-add (L2 atomics)
__global__ void k_edge(const float* __restrict__ H, const int* __restrict__ src,
                       const int* __restrict__ dst, const float* __restrict__ dis,
                       float* __restrict__ agg) {
  int tid = blockIdx.x * blockDim.x + threadIdx.x;
  int e = tid >> 2;
  if (e >= NE) return;
  int c0 = (tid & 3) * 8;
  int s = src[e], d = dst[e];
  float coeff = dis[s] * dis[d];
  const float4* hp = (const float4*)&H[(size_t)s * HC + c0];
  float4 a = hp[0], b = hp[1];
  float* ap = &agg[(size_t)d * HC + c0];
  atomicAdd(ap + 0, a.x * coeff); atomicAdd(ap + 1, a.y * coeff);
  atomicAdd(ap + 2, a.z * coeff); atomicAdd(ap + 3, a.w * coeff);
  atomicAdd(ap + 4, b.x * coeff); atomicAdd(ap + 5, b.y * coeff);
  atomicAdd(ap + 6, b.z * coeff); atomicAdd(ap + 7, b.w * coeff);
}

// ------------ cur(bf16) = tanh(agg + h*dis^2 + b); bf16 matches GEMM staging
__global__ void k_update(const float4* __restrict__ H, const float4* __restrict__ agg,
                         const float* __restrict__ dis, const float* __restrict__ b,
                         __bf16* __restrict__ out) {
  int i = blockIdx.x * blockDim.x + threadIdx.x;      // over NN*HC/4
  if (i >= NN * HC / 4) return;
  int node = i >> 3;
  int c = (i & 7) * 4;
  float d2 = dis[node] * dis[node];
  float4 h4 = H[i], a4 = agg[i];
  float o0 = tanhf(a4.x + h4.x * d2 + b[c + 0]);
  float o1 = tanhf(a4.y + h4.y * d2 + b[c + 1]);
  float o2 = tanhf(a4.z + h4.z * d2 + b[c + 2]);
  float o3 = tanhf(a4.w + h4.w * d2 + b[c + 3]);
  uint2 u; u.x = pack2(o0, o1); u.y = pack2(o2, o3);
  *(uint2*)&out[(size_t)i * 4] = u;
}

// ----------------------------------- per-graph mean pooling (f32 accumulate)
__global__ __launch_bounds__(32) void k_pool(const __bf16* __restrict__ C,
                                             float* __restrict__ hs, int layer) {
  int g = blockIdx.x, c = threadIdx.x;
  float sum = 0.f;
  const __bf16* p = C + (size_t)g * FN * HC + c;
  for (int i = 0; i < FN; ++i) sum += (float)p[i * HC];
  hs[g * (3 * HC) + layer * HC + c] = sum * (1.0f / (float)FN);
}

// ------------------------------------ strict-upper-triangle gather offsets
__global__ void k_triu(int* __restrict__ offs) {
  int i = blockIdx.x, j = threadIdx.x;
  if (j > i) {
    int base = i * 255 - (i * (i - 1)) / 2;
    offs[base + (j - i - 1)] = i * FN + j;
  }
}

// -- fold BN into W0, transpose, convert: Wbf[o][f] = bf16(scale(f)*W0[f][o])
__global__ void k_wscale(const float* __restrict__ W0,
                         const float* __restrict__ bng, const float* __restrict__ bnv,
                         const float* __restrict__ bhg, const float* __restrict__ bhv,
                         __bf16* __restrict__ Wbf) {
  int i = blockIdx.x * blockDim.x + threadIdx.x;      // pairs
  if (i >= IN1 * HID / 2) return;
  int o = i / (IN1 / 2), fp = i - o * (IN1 / 2), f = fp * 2;
  float s0 = (f < TRIU) ? bng[f] * rsqrtf(bnv[f] + EPS)
                        : bhg[f - TRIU] * rsqrtf(bhv[f - TRIU] + EPS);
  int f1 = f + 1;
  float s1 = (f1 < TRIU) ? bng[f1] * rsqrtf(bnv[f1] + EPS)
                         : bhg[f1 - TRIU] * rsqrtf(bhv[f1 - TRIU] + EPS);
  float a = W0[(size_t)f * HID + o] * s0;
  float b = W0[(size_t)f1 * HID + o] * s1;
  *(unsigned*)&Wbf[(size_t)o * IN1 + f] = pack2(a, b);
}

__global__ __launch_bounds__(256) void k_cvec(const float* __restrict__ W0,
    const float* __restrict__ b0,
    const float* __restrict__ bng, const float* __restrict__ bnb,
    const float* __restrict__ bnm, const float* __restrict__ bnv,
    const float* __restrict__ bhg, const float* __restrict__ bhb,
    const float* __restrict__ bhm, const float* __restrict__ bhv,
    float* __restrict__ cvec) {
  __shared__ float red[256];
  int o = blockIdx.x;
  float acc = 0.f;
  for (int f = threadIdx.x; f < IN1; f += 256) {
    float sh;
    if (f < TRIU) sh = bnb[f] - bnm[f] * bng[f] * rsqrtf(bnv[f] + EPS);
    else { int k = f - TRIU; sh = bhb[k] - bhm[k] * bhg[k] * rsqrtf(bhv[k] + EPS); }
    acc += sh * W0[(size_t)f * HID + o];
  }
  red[threadIdx.x] = acc;
  __syncthreads();
  for (int s = 128; s > 0; s >>= 1) {
    if (threadIdx.x < s) red[threadIdx.x] += red[threadIdx.x + s];
    __syncthreads();
  }
  if (threadIdx.x == 0) cvec[o] = b0[o] + red[0];
}

// ------------- MLP layer 0: Z1 = relu(BN0( [triu(x), hs] @ W' + cvec ))
// M=512, N=64, K=32736. 8 waves as 2x4 -> 32 graphs x 64 outs per block.
// TRIU is chunk-aligned: x/hs source select is a UNIFORM per-chunk branch.
__global__ __launch_bounds__(256) void k_mlp_gemm(const float* __restrict__ x,
    const float* __restrict__ hs, const int* __restrict__ offs,
    const __bf16* __restrict__ Wbf, const float* __restrict__ cvec,
    const float* __restrict__ g0, const float* __restrict__ be0,
    const float* __restrict__ m0, const float* __restrict__ v0,
    float* __restrict__ Z1) {
  __shared__ __attribute__((aligned(16))) __bf16 As[2 * 32 * LDA];
  __shared__ __attribute__((aligned(16))) __bf16 Bst[2 * 64 * LDA]; // [out][k]
  const int tid  = threadIdx.x;
  const int wave = tid >> 5;
  const int lane = tid & 31;
  const int half = lane >> 4;
  const int mrow = lane & 15;
  const int mt = wave >> 2, nt = wave & 3;
  const int rowBase = blockIdx.x * 32;

  // per-thread staging geometry (hoisted)
  const int sr = tid >> 3, skq = (tid & 7) * 4;
  const float* xg = x + (size_t)(rowBase + sr) * (FN * FN);
  const float* hg = hs + (rowBase + sr) * (3 * HC);
  const int bcc = tid >> 2, brq = (tid & 3) * 8;

  auto stage = [&](int kc, int buf) {
    __bf16* as = &As[buf * 32 * LDA];
    __bf16* bs = &Bst[buf * 64 * LDA];
    async_copy_b128(lds_addr(&bs[bcc * LDA + brq]),    // B: 64x32 bf16 DMA
                    &Wbf[(size_t)bcc * IN1 + kc + brq]);
    uint2 u;
    if (kc < TRIU) {                                   // uniform branch
      int4 o4 = *(const int4*)&offs[kc + skq];
      u.x = pack2(xg[o4.x], xg[o4.y]);
      u.y = pack2(xg[o4.z], xg[o4.w]);
    } else {
      float4 f4 = *(const float4*)&hg[(kc - TRIU) + skq];
      u.x = pack2(f4.x, f4.y);
      u.y = pack2(f4.z, f4.w);
    }
    *(uint2*)&as[sr * LDA + skq] = u;
  };

  f32x8 c = {};
  const int nc = IN1 / 32;                             // 1023
  stage(0, 0);
  for (int cc = 0; cc < nc; ++cc) {
    wait_async0();
    __syncthreads();
    if (cc + 1 < nc) stage((cc + 1) * 32, (cc + 1) & 1);
    const __bf16* as = &As[(cc & 1) * 32 * LDA];
    const __bf16* bs = &Bst[(cc & 1) * 64 * LDA];

    Frag af, bf;
    const int rb = (mt * 16 + mrow) * LDA;
    const int kb = half * 8;
    af.q[0] = *(const uint4*)&as[rb + kb];
    af.q[1] = *(const uint4*)&as[rb + 16 + kb];
    const int nb = (nt * 16 + mrow) * LDA + half * 8;
    bf.q[0] = *(const uint4*)&bs[nb];
    bf.q[1] = *(const uint4*)&bs[nb + 16];

    c = __builtin_amdgcn_wmma_f32_16x16x32_bf16(false, af.v, false, bf.v,
                                                (short)0, c, false, false);
  }
  const int n = nt * 16 + mrow;
  const float s = g0[n] * rsqrtf(v0[n] + EPS);
#pragma unroll
  for (int v = 0; v < 8; ++v) {
    int gRow = rowBase + mt * 16 + half * 8 + v;
    float z = c[v] + cvec[n];
    z = (z - m0[n]) * s + be0[n];
    Z1[gRow * HID + n] = fmaxf(z, 0.f);
  }
}

// --------------------------- tiny tail MLP: 64 -> 32 -> 32 -> 2 per graph
__global__ __launch_bounds__(64) void k_tail(const float* __restrict__ Z1,
    const float* __restrict__ W1, const float* __restrict__ b1,
    const float* __restrict__ W2, const float* __restrict__ b2,
    const float* __restrict__ W3, const float* __restrict__ b3,
    const float* __restrict__ g1, const float* __restrict__ be1,
    const float* __restrict__ m1, const float* __restrict__ v1,
    const float* __restrict__ g2, const float* __restrict__ be2,
    const float* __restrict__ m2, const float* __restrict__ v2,
    float* __restrict__ out) {
  __shared__ float z[64], z2[32], z3[32];
  int g = blockIdx.x, t = threadIdx.x;
  z[t] = Z1[g * HID + t];
  __syncthreads();
  if (t < 32) {
    float a = b1[t];
    for (int i = 0; i < 64; ++i) a += z[i] * W1[i * 32 + t];
    a = (a - m1[t]) * g1[t] * rsqrtf(v1[t] + EPS) + be1[t];
    z2[t] = fmaxf(a, 0.f);
  }
  __syncthreads();
  if (t < 32) {
    float a = b2[t];
    for (int i = 0; i < 32; ++i) a += z2[i] * W2[i * 32 + t];
    a = (a - m2[t]) * g2[t] * rsqrtf(v2[t] + EPS) + be2[t];
    z3[t] = fmaxf(a, 0.f);
  }
  __syncthreads();
  if (t < 2) {
    float a = b3[t];
    for (int i = 0; i < 32; ++i) a += z3[i] * W3[i * 2 + t];
    out[g * 2 + t] = a;
  }
}

// ---------------------------------------------------------------- launch
extern "C" void kernel_launch(void* const* d_in, const int* in_sizes, int n_in,
                              void* d_out, int out_size, void* d_ws, size_t ws_size,
                              hipStream_t stream) {
  const float* x    = (const float*)d_in[0];
  const int*   ei   = (const int*)d_in[1];
  const int*   srcv = ei;
  const int*   dstv = ei + NE;
  const float* Wc0 = (const float*)d_in[4];  const float* bc0 = (const float*)d_in[5];
  const float* Wc1 = (const float*)d_in[6];  const float* bc1 = (const float*)d_in[7];
  const float* Wc2 = (const float*)d_in[8];  const float* bc2 = (const float*)d_in[9];
  const float* bn_g = (const float*)d_in[10], *bn_b = (const float*)d_in[11],
             * bn_m = (const float*)d_in[12], *bn_v = (const float*)d_in[13];
  const float* bh_g = (const float*)d_in[14], *bh_b = (const float*)d_in[15],
             * bh_m = (const float*)d_in[16], *bh_v = (const float*)d_in[17];
  const float* W0m = (const float*)d_in[18]; const float* b0m = (const float*)d_in[19];
  const float* W1m = (const float*)d_in[20]; const float* b1m = (const float*)d_in[21];
  const float* W2m = (const float*)d_in[22]; const float* b2m = (const float*)d_in[23];
  const float* W3m = (const float*)d_in[24]; const float* b3m = (const float*)d_in[25];
  const float* q0g = (const float*)d_in[26], *q0b = (const float*)d_in[27],
             * q0m = (const float*)d_in[28], *q0v = (const float*)d_in[29];
  const float* q1g = (const float*)d_in[30], *q1b = (const float*)d_in[31],
             * q1m = (const float*)d_in[32], *q1v = (const float*)d_in[33];
  const float* q2g = (const float*)d_in[34], *q2b = (const float*)d_in[35],
             * q2m = (const float*)d_in[36], *q2v = (const float*)d_in[37];
  float* out = (float*)d_out;

  // workspace layout (bytes, 256-aligned regions)
  char* ws = (char*)d_ws;
  float*  dis   = (float*) (ws + 0);                      // NN f32
  float*  bufB  = (float*) (ws + 524288);                 // NN*32 f32 (h)
  float*  agg   = (float*) (ws + 524288 + 16777216);      // NN*32 f32; reused as Wbf
  __bf16* Wbf   = (__bf16*)agg;                           // HID*IN1 bf16 (4.2MB)
  __bf16* curbf = (__bf16*)(ws + 524288 + 2 * 16777216);  // NN*32 bf16 (8MB)
  __bf16* Wct0  = (__bf16*)(ws + 524288 + 2 * 16777216 + 8388608);            // 32*256
  __bf16* Wct1  = Wct0 + 32 * 256;
  __bf16* Wct2  = Wct1 + 32 * 32;
  float*  hs    = (float*) (ws + 524288 + 2 * 16777216 + 8388608 + 32768);    // NG*96
  int*    offs  = (int*)   (ws + 524288 + 2 * 16777216 + 8388608 + 32768 + 196608);
  float*  cvec  = (float*) (ws + 524288 + 2 * 16777216 + 8388608 + 32768 + 196608 + 131072);
  float*  Z1    = (float*) (ws + 524288 + 2 * 16777216 + 8388608 + 32768 + 196608 + 131072 + 256);

  // dis = rsqrt(1 + indeg); conv weights -> transposed bf16
  k_fill4<<<NN / 4 / 256, 256, 0, stream>>>((float4*)dis, 1.0f, NN / 4);
  k_deg<<<NE / 256, 256, 0, stream>>>(dstv, dis);
  k_rsqrt_ip<<<NN / 256, 256, 0, stream>>>(dis, NN);
  k_wconv<<<(32 * 256 + 255) / 256, 256, 0, stream>>>(Wc0, Wct0, 256);
  k_wconv<<<(32 * 32 + 255) / 256, 256, 0, stream>>>(Wc1, Wct1, 32);
  k_wconv<<<(32 * 32 + 255) / 256, 256, 0, stream>>>(Wc2, Wct2, 32);

  const __bf16* convWt[3] = {Wct0, Wct1, Wct2};
  const float*  convB[3]  = {bc0, bc1, bc2};
  for (int l = 0; l < 3; ++l) {
    if (l == 0)
      k_gcn_gemm<false><<<NN / 128, 256, 0, stream>>>(x, nullptr, convWt[0], bufB, FN);
    else
      k_gcn_gemm<true><<<NN / 128, 256, 0, stream>>>(nullptr, curbf, convWt[l], bufB, HC);
    k_fill4<<<NN * HC / 4 / 256, 256, 0, stream>>>((float4*)agg, 0.0f, NN * HC / 4);
    k_edge<<<NE * 4 / 256, 256, 0, stream>>>(bufB, srcv, dstv, dis, agg);
    k_update<<<NN * HC / 4 / 256, 256, 0, stream>>>((const float4*)bufB,
        (const float4*)agg, dis, convB[l], curbf);
    k_pool<<<NG, 32, 0, stream>>>(curbf, hs, l);
  }

  k_triu<<<FN, FN, 0, stream>>>(offs);
  k_wscale<<<(IN1 * HID / 2 + 255) / 256, 256, 0, stream>>>(W0m, bn_g, bn_v,
                                                            bh_g, bh_v, Wbf);
  k_cvec<<<HID, 256, 0, stream>>>(W0m, b0m, bn_g, bn_b, bn_m, bn_v,
                                  bh_g, bh_b, bh_m, bh_v, cvec);
  k_mlp_gemm<<<NG / 32, 256, 0, stream>>>(x, hs, offs, Wbf, cvec,
                                          q0g, q0b, q0m, q0v, Z1);
  k_tail<<<NG, 64, 0, stream>>>(Z1, W1m, b1m, W2m, b2m, W3m, b3m,
                                q1g, q1b, q1m, q1v, q2g, q2b, q2m, q2v, out);
}
